// diff_CSDI_84868553769564
// MI455X (gfx1250) — matrix operand order repaired and checked by reading, not compiled, not run on hardware
//
#include <hip/hip_runtime.h>
#include <hip/hip_bf16.h>
#include <math.h>
#include <stdint.h>

typedef __attribute__((ext_vector_type(16))) _Float16 v16h;
typedef __attribute__((ext_vector_type(8)))  float    v8f;

#define Bb   4
#define KK   14
#define LL   48
#define LP   672      // K*L
#define NN   8
#define SS   64
#define SIDE 144
#define CC   64
#define EMBD 128
#define HEADS 8
#define HD    8

__device__ __forceinline__ int imin(int a, int b) { return a < b ? a : b; }

// ---------------------------------------------------------------------------
// Generic batched, strided WMMA GEMM:
//   Y[z](m,n) = act( alpha * sum_k A[z](m,k) * B[z](k,n) + bias )
// A element: A[aoff(z) + m*lda_m + k]            (A is always k-contiguous here)
// B element: Bm[boff(z) + n*ldb_n + k*ldb_k]
// off(z) = (z / inner)*so + (z % inner)*si  -> handles (batch, head) nests.
// One wave computes one 16x16 tile with v_wmma_f32_16x16x32_f16.
//
// OOB policy: A rows / B cols are CLAMPED (they only feed epilogue-masked
// outputs). The K-tail is peeled into one masked (v_cndmask) iteration.
// BVEC=1: ldb_n == 1; each lane gathers its 16 B values along k at stride ldb_k.
// ---------------------------------------------------------------------------
struct GemmDesc {
  const float* A; const float* Bm; float* Y; const float* bias;
  int M, N, Kd;
  int lda_m, ldb_n, ldb_k, ldy_m, ldy_n;
  long long a_so; int a_si, a_inner;
  long long b_so; int b_si, b_inner;
  long long y_so; int y_si, y_inner;
  float alpha;
  int bias_mode;   // 0 none, 1 per-col(n), 2 per-row(m)
  int act;         // 0 none, 1 relu, 2 exact gelu
};

template<int BVEC>
__global__ __launch_bounds__(32)
void k_wmma_gemm(GemmDesc d) {
  const int lane = threadIdx.x;
  const int hl  = lane >> 4;       // half-wave select
  const int l16 = lane & 15;
  const int m0 = blockIdx.x * 16;
  const int n0 = blockIdx.y * 16;
  const int z  = blockIdx.z;
  const long long aoff = (long long)(z / d.a_inner) * d.a_so + (long long)(z % d.a_inner) * d.a_si;
  const long long boff = (long long)(z / d.b_inner) * d.b_so + (long long)(z % d.b_inner) * d.b_si;
  const long long yoff = (long long)(z / d.y_inner) * d.y_so + (long long)(z % d.y_inner) * d.y_si;

  // clamped A row base (OOB rows feed only masked outputs); rows are contiguous in k
  const int mA = imin(m0 + l16, d.M - 1);
  const float* parow = d.A + aoff + (long long)mA * d.lda_m;
  const int kb_lane = imin(n0 + l16, d.N - 1);  // B fragment column (n) for this lane
  const float* pbbase = d.Bm + boff;
  const int kmain = d.Kd & ~31;

  v8f acc = {};

  if (BVEC) {
    // B column per lane, gathered along k: pointer-increment loop, 4x b128 A loads + 1 wmma per iter
    const float* pb = pbbase + (long long)kb_lane * d.ldb_n;
    const long long bstep = 32LL * d.ldb_k;
#pragma unroll 2
    for (int k0 = 0; k0 < kmain; k0 += 32) {
      const float* pa = parow + k0 + (hl << 3);
      const float4 r0 = *(const float4*)(pa);
      const float4 r1 = *(const float4*)(pa + 4);
      const float4 r2 = *(const float4*)(pa + 16);
      const float4 r3 = *(const float4*)(pa + 20);
      v16h a, b;
      a[0]=(_Float16)r0.x;  a[1]=(_Float16)r0.y;  a[2]=(_Float16)r0.z;  a[3]=(_Float16)r0.w;
      a[4]=(_Float16)r1.x;  a[5]=(_Float16)r1.y;  a[6]=(_Float16)r1.z;  a[7]=(_Float16)r1.w;
      a[8]=(_Float16)r2.x;  a[9]=(_Float16)r2.y;  a[10]=(_Float16)r2.z; a[11]=(_Float16)r2.w;
      a[12]=(_Float16)r3.x; a[13]=(_Float16)r3.y; a[14]=(_Float16)r3.z; a[15]=(_Float16)r3.w;
#pragma unroll
      for (int e = 0; e < 16; ++e) b[e] = (_Float16)pb[(long long)(((e & 8) << 1) + (hl << 3) + (e & 7)) * d.ldb_k];
      acc = __builtin_amdgcn_wmma_f32_16x16x32_f16(false, a, false, b, (short)0, acc, false, false);
      pb += bstep;
    }
  } else {
    // B gather: hoist wave-uniform clamped column offsets out of the loop
    long long noff[16];
#pragma unroll
    for (int e = 0; e < 16; ++e) noff[e] = (long long)imin(n0 + e, d.N - 1) * d.ldb_n;
    const float* pbrow = pbbase + (long long)kb_lane * d.ldb_n;
    const long long bstep = 32LL * d.ldb_k;
    for (int k0 = 0; k0 < kmain; k0 += 32) {
      const float* pa = parow + k0 + (hl << 3);
      const float4 r0 = *(const float4*)(pa);
      const float4 r1 = *(const float4*)(pa + 4);
      const float4 r2 = *(const float4*)(pa + 16);
      const float4 r3 = *(const float4*)(pa + 20);
      v16h a, b;
      a[0]=(_Float16)r0.x;  a[1]=(_Float16)r0.y;  a[2]=(_Float16)r0.z;  a[3]=(_Float16)r0.w;
      a[4]=(_Float16)r1.x;  a[5]=(_Float16)r1.y;  a[6]=(_Float16)r1.z;  a[7]=(_Float16)r1.w;
      a[8]=(_Float16)r2.x;  a[9]=(_Float16)r2.y;  a[10]=(_Float16)r2.z; a[11]=(_Float16)r2.w;
      a[12]=(_Float16)r3.x; a[13]=(_Float16)r3.y; a[14]=(_Float16)r3.z; a[15]=(_Float16)r3.w;
#pragma unroll
      for (int e = 0; e < 16; ++e) b[e] = (_Float16)pbrow[(long long)(((e & 8) << 1) + (hl << 3) + (e & 7)) * d.ldb_k];
      acc = __builtin_amdgcn_wmma_f32_16x16x32_f16(false, a, false, b, (short)0, acc, false, false);
      pbrow += bstep;
    }
  }

  // ---- peeled masked K-tail (Kd % 32 != 0: attention hd=8, cond conv 144) ----
  if (kmain < d.Kd) {
    v16h a, b;
#pragma unroll
    for (int e = 0; e < 16; ++e) {
      const int k = kmain + ((e & 8) << 1) + (hl << 3) + (e & 7);
      const bool ok = k < d.Kd;
      const float v = parow[ok ? k : 0];
      a[e] = (_Float16)(ok ? v : 0.0f);
    }
    const float* pbrow = pbbase + (long long)kb_lane * d.ldb_n;
#pragma unroll
    for (int e = 0; e < 16; ++e) {
      const int kb = kmain + ((e & 8) << 1) + (hl << 3) + (e & 7);
      const bool okb = kb < d.Kd;
      const float v = pbrow[(long long)(okb ? kb : 0) * d.ldb_k];
      b[e] = (_Float16)(okb ? v : 0.0f);
    }
    acc = __builtin_amdgcn_wmma_f32_16x16x32_f16(false, a, false, b, (short)0, acc, false, false);
  }

  // C/D layout: VGPR r -> row m0+r (lanes 0-15) or m0+8+r (lanes 16-31); col = lane&15
#pragma unroll
  for (int r = 0; r < 8; ++r) {
    const int m = m0 + r + hl * 8;
    const int n = n0 + l16;
    if (m < d.M && n < d.N) {
      float v = acc[r] * d.alpha;
      if (d.bias_mode == 1)      v += d.bias[n];
      else if (d.bias_mode == 2) v += d.bias[m];
      if (d.act == 1)      v = fmaxf(v, 0.0f);
      else if (d.act == 2) v = 0.5f * v * (1.0f + erff(v * 0.70710678f));
      d.Y[yoff + (long long)m * d.ldy_m + (long long)n * d.ldy_n] = v;
    }
  }
}

// ------------------------- row softmax (in place) ---------------------------
__global__ __launch_bounds__(128)
void k_softmax(float* x, int ncol) {
  const long long row = blockIdx.x;
  float* p = x + row * (long long)ncol;
  __shared__ float red[128];
  const int t = threadIdx.x;
  float mx = -3.0e38f;
  for (int i = t; i < ncol; i += 128) mx = fmaxf(mx, p[i]);
  red[t] = mx; __syncthreads();
  for (int s = 64; s > 0; s >>= 1) { if (t < s) red[t] = fmaxf(red[t], red[t + s]); __syncthreads(); }
  mx = red[0]; __syncthreads();
  float sm = 0.0f;
  for (int i = t; i < ncol; i += 128) { float e = expf(p[i] - mx); p[i] = e; sm += e; }
  red[t] = sm; __syncthreads();
  for (int s = 64; s > 0; s >>= 1) { if (t < s) red[t] += red[t + s]; __syncthreads(); }
  const float inv = 1.0f / red[0];
  for (int i = t; i < ncol; i += 128) p[i] *= inv;
}

// ------------- layernorm over 64 features: out = LN(a (+ r)) * g (+ beta) ---
// rmode: 0 none, 1 r indexed by row, 2 r broadcast over N (row=(b*N+n)*Lp+s -> r row = b*Lp+s)
__global__ __launch_bounds__(64)
void k_layernorm(const float* a, const float* r, const float* g, const float* beta,
                 float* out, int rmode, int r_div, int r_mod) {
  const long long row = blockIdx.x;
  const int c = threadIdx.x;
  __shared__ float red[64];
  float av = a[row * 64 + c];
  if (rmode == 1) av += r[row * 64 + c];
  else if (rmode == 2) {
    const long long rr = (row / r_div) * (long long)r_mod + (row % r_mod);
    av += r[rr * 64 + c];
  }
  red[c] = av; __syncthreads();
  for (int s = 32; s > 0; s >>= 1) { if (c < s) red[c] += red[c + s]; __syncthreads(); }
  const float mean = red[0] * (1.0f / 64.0f); __syncthreads();
  const float xc = av - mean;
  red[c] = xc * xc; __syncthreads();
  for (int s = 32; s > 0; s >>= 1) { if (c < s) red[c] += red[c + s]; __syncthreads(); }
  const float inv = rsqrtf(red[0] * (1.0f / 64.0f) + 1e-5f);
  float v = xc * inv * g[c];
  if (beta) v += beta[c];
  out[row * 64 + c] = v;
}

// ---------------- diffusion embedding (tiny, one block per sample) ----------
__global__ __launch_bounds__(128)
void k_diffemb(const int* step, const float* w1, const float* b1,
               const float* w2, const float* b2, float* e_out) {
  const int b = blockIdx.x, i = threadIdx.x;
  __shared__ float t0[128], t1[128];
  const float st = (float)step[b];
  if (i < 64) {
    const float fr = powf(10.0f, (float)i * (4.0f / 63.0f));
    t0[i]      = sinf(st * fr);
    t0[i + 64] = cosf(st * fr);
  }
  __syncthreads();
  float acc = b1[i];
  for (int j = 0; j < 128; ++j) acc += w1[i * 128 + j] * t0[j];
  acc = acc / (1.0f + expf(-acc));        // silu
  t1[i] = acc; __syncthreads();
  float acc2 = b2[i];
  for (int j = 0; j < 128; ++j) acc2 += w2[i * 128 + j] * t1[j];
  acc2 = acc2 / (1.0f + expf(-acc2));
  e_out[b * 128 + i] = acc2;
}

// ---------------- GCN normalized adjacency (8x8 per sample) -----------------
__global__ __launch_bounds__(64)
void k_adj(const float* lat, const float* lon, float* Ah) {
  const int b = blockIdx.x, t = threadIdx.x;
  const int i = t >> 3, j = t & 7;
  __shared__ float Am[64]; __shared__ float dinv[8];
  const float dx = lon[b * 8 + i] - lon[b * 8 + j];
  const float dy = lat[b * 8 + i] - lat[b * 8 + j];
  Am[t] = expf(-500.0f * sqrtf(dx * dx + dy * dy));
  __syncthreads();
  if (t < 8) { float s = 0.0f; for (int q = 0; q < 8; ++q) s += Am[t * 8 + q]; dinv[t] = rsqrtf(s); }
  __syncthreads();
  Ah[b * 64 + t] = dinv[i] * Am[t] * dinv[j];
}

// ------------------------------ elementwise ---------------------------------
__global__ void k_zero(float* p, long long n) {
  const long long i = (long long)blockIdx.x * 256 + threadIdx.x;
  if (i < n) p[i] = 0.0f;
}
__global__ void k_inconv(const float* x, const float* w, const float* b, float* h) {
  const int idx = blockIdx.x * 256 + threadIdx.x;
  if (idx >= Bb * CC * LP) return;
  const int bq = idx / (CC * LP), rem = idx % (CC * LP), c = rem / LP, lp = rem % LP;
  h[idx] = fmaxf(w[c] * x[bq * LP + lp] + b[c], 0.0f);
}
__global__ void k_add_dp(const float* h, const float* dp, float* y) {
  const int idx = blockIdx.x * 256 + threadIdx.x;
  if (idx >= Bb * CC * LP) return;
  const int bq = idx / (CC * LP), c = (idx % (CC * LP)) / LP;
  y[idx] = h[idx] + dp[bq * CC + c];
}
// token-major gather/scatter between y[b,c,lp] and X[t,c], t=s*NB+nb, nb=b*inner+i, lp=s*ps+i*pi
__global__ void k_tok_gather(const float* y, float* X, int T, int NB, int inner, int ps, int pi) {
  const int idx = blockIdx.x * 256 + threadIdx.x;
  if (idx >= T * 64) return;
  const int t = idx >> 6, c = idx & 63;
  const int s = t / NB, nb = t % NB;
  const int b = nb / inner, i = nb % inner;
  X[idx] = y[(long long)b * (CC * LP) + (long long)c * LP + s * ps + i * pi];
}
__global__ void k_tok_scatter(float* y, const float* X, int T, int NB, int inner, int ps, int pi) {
  const int idx = blockIdx.x * 256 + threadIdx.x;
  if (idx >= T * 64) return;
  const int t = idx >> 6, c = idx & 63;
  const int s = t / NB, nb = t % NB;
  const int b = nb / inner, i = nb % inner;
  y[(long long)b * (CC * LP) + (long long)c * LP + s * ps + i * pi] = X[idx];
}
__global__ void k_transpose_cl(const float* y, float* y2) {  // [b,c,lp] -> [b,lp,c]
  const int idx = blockIdx.x * 256 + threadIdx.x;
  if (idx >= Bb * CC * LP) return;
  const int b = idx / (CC * LP), rem = idx % (CC * LP), c = rem / LP, lp = rem % LP;
  y2[(long long)b * (CC * LP) + (long long)lp * CC + c] = y[idx];
}
__global__ void k_mean_add(const float* go, float* y2) {  // y2[b,lp,c] += mean_n go[b,n,lp*64+c]
  const int idx = blockIdx.x * 256 + threadIdx.x;
  if (idx >= Bb * LP * CC) return;
  const int b = idx / (LP * CC), r = idx % (LP * CC);
  float s = 0.0f;
  for (int n = 0; n < NN; ++n) s += go[(long long)b * NN * LP * CC + (long long)n * LP * CC + r];
  y2[idx] += 0.125f * s;
}
__global__ void k_gate(const float* mid, const float* cnd, float* gated) {
  const int idx = blockIdx.x * 256 + threadIdx.x;
  if (idx >= Bb * CC * LP) return;
  const int b = idx / (CC * LP), rem = idx % (CC * LP), c = rem / LP, lp = rem % LP;
  const long long base = (long long)b * 2 * CC * LP;
  const float g = mid[base + (long long)c * LP + lp]        + cnd[base + (long long)c * LP + lp];
  const float f = mid[base + (long long)(c + CC) * LP + lp] + cnd[base + (long long)(c + CC) * LP + lp];
  gated[idx] = (1.0f / (1.0f + expf(-g))) * tanhf(f);
}
__global__ void k_resskip(const float* oo, float* h, float* skip) {
  const int idx = blockIdx.x * 256 + threadIdx.x;
  if (idx >= Bb * CC * LP) return;
  const int b = idx / (CC * LP), rem = idx % (CC * LP), c = rem / LP, lp = rem % LP;
  const long long base = (long long)b * 2 * CC * LP;
  h[idx] = (h[idx] + oo[base + (long long)c * LP + lp]) * 0.70710678f;
  skip[idx] += oo[base + (long long)(c + CC) * LP + lp];
}
__global__ void k_outconv2(const float* o1, const float* w, const float* b2, float* out) {
  const int idx = blockIdx.x * 256 + threadIdx.x;
  if (idx >= Bb * LP) return;
  const int b = idx / LP, lp = idx % LP;
  float s = b2[0];
  for (int c = 0; c < CC; ++c) s += o1[(long long)b * CC * LP + (long long)c * LP + lp] * w[c];
  out[idx] = s;
}

// ----------------------------- host helpers ---------------------------------
static inline void launch_gemm(hipStream_t st,
    const float* A, int lda_m, int lda_k, long long a_so, int a_si, int a_inner,
    const float* Bm, int ldb_n, int ldb_k, long long b_so, int b_si, int b_inner,
    float* Y, int ldy_m, int ldy_n, long long y_so, int y_si, int y_inner,
    const float* bias, int bias_mode, int act, float alpha,
    int M, int N, int Kd, int batch) {
  (void)lda_k;  // A is k-contiguous in every call of this network
  GemmDesc d;
  d.A = A; d.Bm = Bm; d.Y = Y; d.bias = bias;
  d.M = M; d.N = N; d.Kd = Kd;
  d.lda_m = lda_m; d.ldb_n = ldb_n; d.ldb_k = ldb_k;
  d.ldy_m = ldy_m; d.ldy_n = ldy_n;
  d.a_so = a_so; d.a_si = a_si; d.a_inner = a_inner < 1 ? 1 : a_inner;
  d.b_so = b_so; d.b_si = b_si; d.b_inner = b_inner < 1 ? 1 : b_inner;
  d.y_so = y_so; d.y_si = y_si; d.y_inner = y_inner < 1 ? 1 : y_inner;
  d.alpha = alpha; d.bias_mode = bias_mode; d.act = act;
  dim3 grid((M + 15) / 16, (N + 15) / 16, batch);
  const bool bvec = (ldb_n == 1) && ((ldb_k & 3) == 0) && ((b_so & 3) == 0) &&
                    ((b_si & 3) == 0) && ((N & 15) == 0) &&
                    ((((uintptr_t)Bm) & 15) == 0);
  if (bvec) k_wmma_gemm<1><<<grid, 32, 0, st>>>(d);
  else      k_wmma_gemm<0><<<grid, 32, 0, st>>>(d);
}

struct EncP { const float *f1_b,*f1_w,*f2_b,*f2_w,*in_b,*in_w,*ln1_b,*ln1_g,*ln2_b,*ln2_g,*o_b,*o_w; };
struct DecP { const float *ca_in_w,*ca_o_w,*f1_w,*f2_w,*ln1_g,*ln2_g,*ln3_g,*sa_in_w,*sa_o_w; };
struct LayP { const float *cond_b,*cond_w,*diff_b,*diff_w,*gcn_b1,*gcn_b3,*gcn_w1,*gcn_w3,
                          *mid_b,*mid_w,*out_b,*out_w; DecP dec; EncP feat, time; };

static EncP get_enc(void* const* din, int base) {
  EncP p;
  p.f1_b=(const float*)din[base+0];  p.f1_w=(const float*)din[base+1];
  p.f2_b=(const float*)din[base+2];  p.f2_w=(const float*)din[base+3];
  p.in_b=(const float*)din[base+4];  p.in_w=(const float*)din[base+5];
  p.ln1_b=(const float*)din[base+6]; p.ln1_g=(const float*)din[base+7];
  p.ln2_b=(const float*)din[base+8]; p.ln2_g=(const float*)din[base+9];
  p.o_b=(const float*)din[base+10];  p.o_w=(const float*)din[base+11];
  return p;
}
static DecP get_dec(void* const* din, int base) {
  DecP p;
  p.ca_in_w=(const float*)din[base+0]; p.ca_o_w=(const float*)din[base+1];
  p.f1_w=(const float*)din[base+2];    p.f2_w=(const float*)din[base+3];
  p.ln1_g=(const float*)din[base+4];   p.ln2_g=(const float*)din[base+5];
  p.ln3_g=(const float*)din[base+6];   p.sa_in_w=(const float*)din[base+7];
  p.sa_o_w=(const float*)din[base+8];
  return p;
}
static LayP get_layer(void* const* din, int base) {  // leaves sorted by key per jax pytree order
  LayP p;
  p.cond_b=(const float*)din[base+0];  p.cond_w=(const float*)din[base+1];
  p.dec = get_dec(din, base + 2);                       // 9 leaves
  p.diff_b=(const float*)din[base+11]; p.diff_w=(const float*)din[base+12];
  p.feat = get_enc(din, base + 13);                     // 12 leaves
  p.gcn_b1=(const float*)din[base+25]; p.gcn_b3=(const float*)din[base+26];
  p.gcn_w1=(const float*)din[base+27]; p.gcn_w3=(const float*)din[base+28];
  p.mid_b=(const float*)din[base+29];  p.mid_w=(const float*)din[base+30];
  p.out_b=(const float*)din[base+31];  p.out_w=(const float*)din[base+32];
  p.time = get_enc(din, base + 33);                     // 12 leaves
  return p;
}

// Full encoder layer on token-major X[T,64]; T = S*NB, token t = s*NB + nb.
static void run_encoder(hipStream_t st, float* X, float* qkv, float* logits,
                        float* t1, float* t2, int S, int NB, const EncP& p) {
  const int T = S * NB;
  // QKV projection
  launch_gemm(st, X,64,1, 0,0,1,  p.in_w,64,1, 0,0,1,  qkv,192,1, 0,0,1,
              p.in_b,1,0, 1.0f, T,192,64, 1);
  // logits[z=nb*H+h][sq][sk] = (Q.K^T)/sqrt(hd)
  launch_gemm(st, qkv,    NB*192,1, 192,8,HEADS,
                  qkv+64, NB*192,1, 192,8,HEADS,
                  logits, S,1, (long long)S*S,0,1,
              nullptr,0,0, 0.35355339f, S,S,HD, NB*HEADS);
  k_softmax<<<NB*HEADS*S,128,0,st>>>(logits, S);
  // O = A.V scattered back token-major into t1
  launch_gemm(st, logits, S,1, (long long)S*S,0,1,
                  qkv+128, 1,NB*192, 192,8,HEADS,
                  t1, NB*64,1, 64,8,HEADS,
              nullptr,0,0, 1.0f, S,HD,S, NB*HEADS);
  // out projection
  launch_gemm(st, t1,64,1, 0,0,1, p.o_w,64,1, 0,0,1, t2,64,1, 0,0,1,
              p.o_b,1,0, 1.0f, T,64,64, 1);
  k_layernorm<<<T,64,0,st>>>(X, t2, p.ln1_g, p.ln1_b, X, 1, 1, 1);
  // FF: gelu(X@f1^T+b1)@f2^T+b2
  launch_gemm(st, X,64,1, 0,0,1, p.f1_w,64,1, 0,0,1, t1,64,1, 0,0,1, p.f1_b,1,2, 1.0f, T,64,64, 1);
  launch_gemm(st, t1,64,1, 0,0,1, p.f2_w,64,1, 0,0,1, t2,64,1, 0,0,1, p.f2_b,1,0, 1.0f, T,64,64, 1);
  k_layernorm<<<T,64,0,st>>>(X, t2, p.ln2_g, p.ln2_b, X, 1, 1, 1);
}

// ---------------------------------------------------------------------------
extern "C" void kernel_launch(void* const* d_in, const int* in_sizes, int n_in,
                              void* d_out, int out_size, void* d_ws, size_t ws_size,
                              hipStream_t stream) {
  (void)in_sizes; (void)n_in; (void)out_size; (void)ws_size;
  const float* x_in = (const float*)d_in[0];
  const float* enc  = (const float*)d_in[1];
  const float* cond = (const float*)d_in[2];
  const int*   dstp = (const int*)d_in[3];
  const float* lat  = (const float*)d_in[4];
  const float* lon  = (const float*)d_in[5];
  const float* emb1_b=(const float*)d_in[6];  const float* emb1_w=(const float*)d_in[7];
  const float* emb2_b=(const float*)d_in[8];  const float* emb2_w=(const float*)d_in[9];
  const float* in_b =(const float*)d_in[10];  const float* in_w =(const float*)d_in[11];
  const float* out1_b=(const float*)d_in[102]; const float* out1_w=(const float*)d_in[103];
  const float* out2_b=(const float*)d_in[104]; const float* out2_w=(const float*)d_in[105];

  // scratch arena (floats)
  float* w = (float*)d_ws; long long off = 0;
  auto alloc = [&](long long n) { float* p = w + off; off += n; return p; };
  const long long BCL = (long long)Bb * CC * LP;            // 172032
  float* h_buf = alloc(BCL);
  float* skip  = alloc(BCL);
  float* ybuf  = alloc(BCL);
  float* y2    = alloc(BCL);
  float* e_emb = alloc(Bb * EMBD);
  float* dp    = alloc(Bb * CC);
  float* tokx  = alloc(BCL);
  float* tok1  = alloc(BCL);
  float* tok2  = alloc(BCL);
  float* qkv   = alloc((long long)2688 * 192);
  float* logits= alloc((long long)HEADS * LP * LP);          // 3.6M floats, reused per stage
  float* saqkv = alloc((long long)Bb * LP * 192);
  float* sao   = alloc(BCL);
  float* dtmp  = alloc(BCL);
  float* xsa   = alloc(BCL);
  float* qx    = alloc(BCL);
  float* kmem  = alloc((long long)Bb * NN * SS * CC);
  float* vmem  = alloc((long long)Bb * NN * SS * CC);
  float* cao   = alloc((long long)Bb * NN * LP * CC);
  float* x2    = alloc((long long)Bb * NN * LP * CC);
  float* ffb   = alloc((long long)Bb * NN * LP * CC);
  float* feats = alloc((long long)Bb * NN * LP * CC);
  float* t3    = alloc((long long)Bb * NN * LP * CC);
  float* gobuf = alloc((long long)Bb * NN * LP * CC);
  float* AhB   = alloc(Bb * 64);
  float* g1    = alloc(Bb * NN * 32);
  float* h1    = alloc(Bb * NN * 32);
  float* midb  = alloc((long long)Bb * 2 * CC * LP);
  float* condo = alloc((long long)Bb * 2 * CC * LP);
  float* gated = alloc(BCL);
  float* oo    = alloc((long long)Bb * 2 * CC * LP);

  const int EG = (int)((BCL + 255) / 256);

  k_zero<<<EG,256,0,stream>>>(skip, BCL);
  k_inconv<<<EG,256,0,stream>>>(x_in, in_w, in_b, h_buf);
  k_diffemb<<<Bb,128,0,stream>>>(dstp, emb1_w, emb1_b, emb2_w, emb2_b, e_emb);
  k_adj<<<Bb,64,0,stream>>>(lat, lon, AhB);

  for (int li = 0; li < 2; ++li) {
    LayP P = get_layer(d_in, 12 + li * 45);
    // diffusion projection [B,64] and broadcast add
    launch_gemm(stream, e_emb,128,1, 0,0,1, P.diff_w,128,1, 0,0,1, dp,64,1, 0,0,1,
                P.diff_b,1,0, 1.0f, Bb,CC,EMBD, 1);
    k_add_dp<<<EG,256,0,stream>>>(h_buf, dp, ybuf);

    // ---- time attention: seq=L, batch=B*K ----
    k_tok_gather<<<EG,256,0,stream>>>(ybuf, tokx, 2688, Bb*KK, KK, 1, LL);
    run_encoder(stream, tokx, qkv, logits, tok1, tok2, LL, Bb*KK, P.time);
    k_tok_scatter<<<EG,256,0,stream>>>(ybuf, tokx, 2688, Bb*KK, KK, 1, LL);
    // ---- feature attention: seq=K, batch=B*L ----
    k_tok_gather<<<EG,256,0,stream>>>(ybuf, tokx, 2688, Bb*LL, LL, LL, 1);
    run_encoder(stream, tokx, qkv, logits, tok1, tok2, KK, Bb*LL, P.feat);
    k_tok_scatter<<<EG,256,0,stream>>>(ybuf, tokx, 2688, Bb*LL, LL, LL, 1);

    k_transpose_cl<<<EG,256,0,stream>>>(ybuf, y2);          // y2[b,lp,c]

    // ---- decoder self-attn (tgt identical over N -> compute once per b) ----
    launch_gemm(stream, y2,64,1, 43008,0,1, P.dec.sa_in_w,64,1, 0,0,1,
                saqkv,192,1, 129024,0,1, nullptr,0,0, 1.0f, LP,192,64, Bb);
    for (int b = 0; b < Bb; ++b) {
      const float* qb = saqkv + (long long)b * 129024;
      launch_gemm(stream, qb,192,1, 8,0,1, qb+64,192,1, 8,0,1,
                  logits, LP,1, (long long)LP*LP,0,1,
                  nullptr,0,0, 0.35355339f, LP,LP,HD, HEADS);
      k_softmax<<<HEADS*LP,128,0,stream>>>(logits, LP);
      launch_gemm(stream, logits, LP,1, (long long)LP*LP,0,1,
                  qb+128, 1,192, 8,0,1,
                  sao + (long long)b*43008, 64,1, 8,0,1,
                  nullptr,0,0, 1.0f, LP,HD,LP, HEADS);
    }
    launch_gemm(stream, sao,64,1, 43008,0,1, P.dec.sa_o_w,64,1, 0,0,1,
                dtmp,64,1, 43008,0,1, nullptr,0,0, 1.0f, LP,64,64, Bb);
    k_layernorm<<<Bb*LP,64,0,stream>>>(y2, dtmp, P.dec.ln1_g, nullptr, xsa, 1, 1, 1);

    // ---- decoder cross-attn ----
    launch_gemm(stream, xsa,64,1, 43008,0,1, P.dec.ca_in_w,64,1, 0,0,1,
                qx,64,1, 43008,0,1, nullptr,0,0, 1.0f, LP,64,64, Bb);
    launch_gemm(stream, enc,64,1, 4096,0,1, P.dec.ca_in_w+4096,64,1, 0,0,1,
                kmem,64,1, 4096,0,1, nullptr,0,0, 1.0f, SS,64,64, Bb*NN);
    launch_gemm(stream, enc,64,1, 4096,0,1, P.dec.ca_in_w+8192,64,1, 0,0,1,
                vmem,64,1, 4096,0,1, nullptr,0,0, 1.0f, SS,64,64, Bb*NN);
    for (int b = 0; b < Bb; ++b) {
      launch_gemm(stream, qx + (long long)b*43008, 64,1, 0,8,HEADS,     // q shared over n
                  kmem + (long long)b*NN*4096, 64,1, 4096,8,HEADS,
                  logits, 64,1, 43008,0,1,
                  nullptr,0,0, 0.35355339f, LP,SS,HD, NN*HEADS);
      k_softmax<<<NN*HEADS*LP,128,0,stream>>>(logits, SS);
      launch_gemm(stream, logits, 64,1, 43008,0,1,
                  vmem + (long long)b*NN*4096, 1,64, 4096,8,HEADS,
                  cao + (long long)b*NN*43008, 64,1, 43008,8,HEADS,
                  nullptr,0,0, 1.0f, LP,HD,SS, NN*HEADS);
    }
    launch_gemm(stream, cao,64,1, 43008,0,1, P.dec.ca_o_w,64,1, 0,0,1,
                ffb,64,1, 43008,0,1, nullptr,0,0, 1.0f, LP,64,64, Bb*NN);
    k_layernorm<<<Bb*NN*LP,64,0,stream>>>(ffb, xsa, P.dec.ln2_g, nullptr, x2, 2, NN*LP, LP);
    // FF + LN3
    launch_gemm(stream, x2,64,1, 0,0,1, P.dec.f1_w,64,1, 0,0,1, ffb,64,1, 0,0,1,
                nullptr,0,2, 1.0f, Bb*NN*LP,64,64, 1);
    launch_gemm(stream, ffb,64,1, 0,0,1, P.dec.f2_w,64,1, 0,0,1, gobuf,64,1, 0,0,1,
                nullptr,0,0, 1.0f, Bb*NN*LP,64,64, 1);
    k_layernorm<<<Bb*NN*LP,64,0,stream>>>(gobuf, x2, P.dec.ln3_g, nullptr, feats, 1, 1, 1);

    // ---- GCN ----
    launch_gemm(stream, feats,43008,1, 344064,0,1, P.gcn_w1, 1,32, 0,0,1,
                g1,32,1, 256,0,1, nullptr,0,0, 1.0f, NN,32,43008, Bb);
    launch_gemm(stream, AhB,8,1, 64,0,1, g1, 1,32, 256,0,1,
                h1,32,1, 256,0,1, P.gcn_b1,1,1, 1.0f, NN,32,NN, Bb);
    launch_gemm(stream, h1,32,1, 256,0,1, P.gcn_w3, 1,43008, 0,0,1,
                t3,43008,1, 344064,0,1, nullptr,0,0, 1.0f, NN,43008,32, Bb);
    launch_gemm(stream, AhB,8,1, 64,0,1, t3, 1,43008, 344064,0,1,
                gobuf,43008,1, 344064,0,1, P.gcn_b3,1,0, 1.0f, NN,43008,NN, Bb);
    k_mean_add<<<EG,256,0,stream>>>(gobuf, y2);

    // ---- mid/cond conv + gate + out conv + residual/skip ----
    launch_gemm(stream, P.mid_w,64,1, 0,0,1, y2, 64,1, 43008,0,1,
                midb,672,1, 86016,0,1, P.mid_b,2,0, 1.0f, 2*CC,LP,CC, Bb);
    launch_gemm(stream, P.cond_w,144,1, 0,0,1, cond, 1,672, 96768,0,1,
                condo,672,1, 86016,0,1, P.cond_b,2,0, 1.0f, 2*CC,LP,SIDE, Bb);
    k_gate<<<EG,256,0,stream>>>(midb, condo, gated);
    launch_gemm(stream, P.out_w,64,1, 0,0,1, gated, 1,672, 43008,0,1,
                oo,672,1, 86016,0,1, P.out_b,2,0, 1.0f, 2*CC,LP,CC, Bb);
    k_resskip<<<EG,256,0,stream>>>(oo, h_buf, skip);
  }

  // output head: relu(conv1(skip/sqrt(2))) then conv to 1 channel
  launch_gemm(stream, out1_w,64,1, 0,0,1, skip, 1,672, 43008,0,1,
              gated,672,1, 43008,0,1, out1_b,2,1, 0.70710678f, CC,LP,CC, Bb);
  k_outconv2<<<(Bb*LP + 255)/256,256,0,stream>>>(gated, out2_w, out2_b, (float*)d_out);
}
